// Estor_45595372814585
// MI455X (gfx1250) — matrix operand order repaired and checked by reading, not compiled, not run on hardware
//
#include <hip/hip_runtime.h>
#include <math.h>

// ---------------------------------------------------------------------------
// MI455X (gfx1250) implementation. Compute-bound (~320 GFLOP vs ~15us of HBM
// time at 23.3 TB/s), so all GEMMs run through v_wmma_f32_16x16x32_bf16.
// GEMM main loops stage tiles with GLOBAL_LOAD_ASYNC_TO_LDS_B128 (ASYNCcnt)
// into double-buffered LDS (BK=64 -> 16 WMMAs per pipeline stage, two
// barriers per 16 WMMAs), overlapping global latency with WMMA issue.
// All x@W.T GEMMs map natively onto the CDNA5 B-fragment layout (B column n
// contiguous in K == W row n), so no transposes are needed except V (staged
// transposed in LDS inside the attention kernel).
// ---------------------------------------------------------------------------

typedef __bf16 bf16;
typedef __attribute__((ext_vector_type(16))) __bf16 v16bf;
typedef __attribute__((ext_vector_type(8)))  __bf16 v8bf;
typedef __attribute__((ext_vector_type(8)))  float  v8f;

#define DEV __device__ __forceinline__

constexpr int Bb     = 8;
constexpr int S_LEN  = 2048;
constexpr int Hh     = 768;
constexpr int NHh    = 12;
constexpr int NLl    = 9;
constexpr int I_FF   = 3072;
constexpr int SPAN   = 64;
constexpr int NSP    = S_LEN / SPAN;     // 32
constexpr int DHh    = Hh / NHh;         // 64
constexpr int NSPANS = Bb * NSP;         // 256
constexpr int M_TOK  = Bb * S_LEN;       // 16384
constexpr int RPS_AT = SPAN + 1;         // 65
constexpr int M_AT   = NSPANS * RPS_AT;  // 16640
constexpr int QKVW   = 3 * Hh;           // 2304

// ---------------- WMMA helpers (wave32 fragment layouts, ISA 7.12.2) -------

DEV v8f zero8() {
  v8f z;
#pragma unroll
  for (int i = 0; i < 8; ++i) z[i] = 0.f;
  return z;
}

DEV v8f wmma_bf16(v16bf a, v16bf b, v8f c) {
  return __builtin_amdgcn_wmma_f32_16x16x32_bf16(false, a, false, b,
                                                 (short)0, c, false, false);
}

// A fragment (16x32, row-major, K-contiguous). rowbase = &tile[row*stride + k0].
// lane<16: K {0..7, 16..23}; lane>=16: K {8..15, 24..31}. Two b128 LDS loads.
DEV v16bf frag_a_ld(const bf16* rowbase) {
  const int half8 = ((threadIdx.x >> 4) & 1) * 8;
  v8bf lo = *(const v8bf*)(rowbase + half8);
  v8bf hi = *(const v8bf*)(rowbase + half8 + 16);
  v16bf r;
#pragma unroll
  for (int i = 0; i < 8; ++i) { r[i] = lo[i]; r[i + 8] = hi[i]; }
  return r;
}

// B fragment (32x16). Column n of B stored as row n of an [N][K] tile.
// lane<16: K {0..7, 16..23}; lane>=16: K {8..15, 24..31}. rowbase = &tile[n*stride + k0].
DEV v16bf frag_b_ld(const bf16* rowbase) {
  const int half8 = ((threadIdx.x >> 4) & 1) * 8;
  v8bf lo = *(const v8bf*)(rowbase + half8);
  v8bf hi = *(const v8bf*)(rowbase + half8 + 16);
  v16bf r;
#pragma unroll
  for (int i = 0; i < 8; ++i) { r[i] = lo[i]; r[i + 8] = hi[i]; }
  return r;
}

// Async global->LDS 16B copy (per-lane addresses), tracked by ASYNCcnt.
DEV void async_cp16(unsigned int lds_off, const void* gaddr) {
  asm volatile("global_load_async_to_lds_b128 %0, %1, off"
               :: "v"(lds_off), "v"(gaddr)
               : "memory");
}

DEV float wave_sum(float v) {
#pragma unroll
  for (int o = 16; o; o >>= 1) v += __shfl_xor(v, o, 32);
  return v;
}
DEV float wave_max(float v) {
#pragma unroll
  for (int o = 16; o; o >>= 1) v = fmaxf(v, __shfl_xor(v, o, 32));
  return v;
}

// ---------------- small elementwise kernels --------------------------------

__global__ void f2bf_kernel(const float* __restrict__ in, bf16* __restrict__ out, int n) {
  int i = blockIdx.x * 256 + threadIdx.x;
  if (i < n) out[i] = (bf16)in[i];
}

// Interleaved-pair RoPE over the full sequence, fp32 -> bf16.
__global__ void rope_kernel(const float* __restrict__ we, bf16* __restrict__ xe) {
  const int HALF = Hh / 2;  // 384
  size_t p = (size_t)blockIdx.x * 256 + threadIdx.x;
  if (p >= (size_t)M_TOK * HALF) return;
  int h = (int)(p % HALF);
  size_t row = p / HALF;
  int pos = (int)(row % S_LEN);
  float inv = __expf(-(float)h * (logf(10000.f) / (float)HALF));
  float ang = (float)pos * inv;
  float s, c;
  __sincosf(ang, &s, &c);
  float a = we[row * Hh + 2 * h];
  float b = we[row * Hh + 2 * h + 1];
  xe[row * Hh + 2 * h]     = (bf16)(a * c - b * s);
  xe[row * Hh + 2 * h + 1] = (bf16)(a * s + b * c);
}

// Write tag_emb[span_tags[s]] into slot 0 of each 65-token span.
__global__ void tag_fill_kernel(const int* __restrict__ tags,
                                const float* __restrict__ temb,
                                bf16* __restrict__ Xat) {
  int s = blockIdx.x;
  int tg = tags[s];
  for (int k = threadIdx.x; k < Hh; k += 256)
    Xat[(size_t)s * RPS_AT * Hh + k] = (bf16)temb[(size_t)tg * Hh + k];
}

// ---------------- generic WMMA GEMM:  Out = act(A @ W.T + bias [+resid]) ---
// A: [M,K] bf16 row-major.  W: [N,K] bf16 row-major (native for x@W.T).
// remap: 0 none; 1 output row -> span*65 + tok + 1; 2 A row -> span*65 + tok + 1.
// BM=128 x BN=128 x BK=64, 8 waves, 16 WMMA tiles per wave per k-step.
// Tiles staged with async global->LDS into a double buffer and pipelined.
__global__ void gemm_bf16_kernel(const bf16* __restrict__ A,
                                 const bf16* __restrict__ W,
                                 const float* __restrict__ bias,
                                 const bf16* __restrict__ resid,
                                 bf16* __restrict__ Out,
                                 int M, int N, int K, int relu, int remap) {
  constexpr int BK = 64;
  constexpr int TILE_E = 128 * BK;  // elements per A or B tile
  __shared__ alignas(16) bf16 tiles[2][2][TILE_E];  // [buf][A|B], 64 KB

  const int m0 = blockIdx.y * 128;
  const int n0 = blockIdx.x * 128;
  const int w = threadIdx.x >> 5, lane = threadIdx.x & 31;
  const int lr = lane & 15, half = lane >> 4;
  const int mt0 = (w & 3) * 2;     // first of 2 M-tiles owned by this wave
  const int nbase = (w >> 2) * 4;  // first of 4 N-tiles owned by this wave

  const unsigned int ldsbase = (unsigned int)(size_t)&tiles[0][0][0];

  // Issue 8 async 16B copies per thread for tile kt into buffer buf.
  auto stage = [&](int kt, int buf) {
#pragma unroll
    for (int pp = 0; pp < 4; ++pp) {
      int idx = threadIdx.x + pp * 256;
      int row = idx >> 3;
      int cg = (idx & 7) * 8;
      int mlog = m0 + row;
      int arow = (remap == 2) ? ((mlog >> 6) * RPS_AT + (mlog & 63) + 1) : mlog;
      const bf16* ga = A + (size_t)arow * K + kt * BK + cg;
      unsigned int la =
          ldsbase + (unsigned int)(((buf * 2 + 0) * TILE_E + row * BK + cg) * 2);
      async_cp16(la, ga);
      const bf16* gb = W + (size_t)(n0 + row) * K + kt * BK + cg;
      unsigned int lb =
          ldsbase + (unsigned int)(((buf * 2 + 1) * TILE_E + row * BK + cg) * 2);
      async_cp16(lb, gb);
    }
  };

  v8f acc0[4], acc1[4];
#pragma unroll
  for (int j = 0; j < 4; ++j) { acc0[j] = zero8(); acc1[j] = zero8(); }

  const int nk = K >> 6;
  stage(0, 0);
  for (int kt = 0; kt < nk; ++kt) {
    int nxt = (kt + 1 < nk) ? kt + 1 : kt;       // clamp: last stage redundant
    stage(nxt, (kt + 1) & 1);
    // Async loads complete in order: <=8 outstanding => tile kt has landed.
    asm volatile("s_wait_asynccnt 0x8" ::: "memory");
    __syncthreads();

    const bf16* Ab = &tiles[kt & 1][0][0];
    const bf16* Bbuf = &tiles[kt & 1][1][0];
#pragma unroll
    for (int ks = 0; ks < BK; ks += 32) {
      v16bf a0 = frag_a_ld(&Ab[(mt0 * 16 + lr) * BK + ks]);
      v16bf a1 = frag_a_ld(&Ab[((mt0 + 1) * 16 + lr) * BK + ks]);
#pragma unroll
      for (int j = 0; j < 4; ++j) {
        v16bf b = frag_b_ld(&Bbuf[((nbase + j) * 16 + lr) * BK + ks]);
        acc0[j] = wmma_bf16(a0, b, acc0[j]);
        acc1[j] = wmma_bf16(a1, b, acc1[j]);
      }
    }
    __syncthreads();  // protect buffer (kt&1) before it is restaged at kt+1
  }

  // epilogue: lane L -> N = L%16, element i -> M = i + 8*(L/16)
#pragma unroll
  for (int j = 0; j < 4; ++j) {
    int n = n0 + (nbase + j) * 16 + lr;
    float bv = bias ? bias[n] : 0.f;
#pragma unroll
    for (int hm = 0; hm < 2; ++hm) {
      const v8f& a = hm ? acc1[j] : acc0[j];
      int mt = mt0 + hm;
#pragma unroll
      for (int i = 0; i < 8; ++i) {
        int mlog = m0 + mt * 16 + i + half * 8;
        float v = a[i] + bv;
        if (resid) v += (float)resid[(size_t)mlog * N + n];
        if (relu) v = fmaxf(v, 0.f);
        int orow = (remap == 1) ? ((mlog >> 6) * RPS_AT + (mlog & 63) + 1) : mlog;
        Out[(size_t)orow * N + n] = (bf16)v;
      }
    }
  }
}

// ---------------- fused per-(span, head) attention -------------------------
// QKV: [nspans*rps, 2304] bf16 rows (Q|K|V). One block per (span, head).
// T tokens (64 or 65), padded: TP = rnd16(T) (<=80), KP = rnd32(T) (<=96).
__global__ void span_attn_kernel(const bf16* __restrict__ QKV,
                                 bf16* __restrict__ Oout,
                                 int T, int rps) {
  constexpr int SP = 80;  // scores stride
  constexpr int KM = 96;  // P / Vt stride
  __shared__ alignas(16) float Ssc[80 * SP];       // 25.6 KB
  __shared__ alignas(16) bf16 Vt[64 * KM];         // 12.3 KB (V transposed)
  __shared__ alignas(16) bf16 QK[2 * 80 * 64];     // 20.5 KB; P overlays this
  bf16* Qs = QK;
  bf16* Ks = QK + 80 * 64;
  bf16* Ps = QK;  // 80*96 bf16 = 15.4 KB <= 20.5 KB, used after Q/K are dead

  const int span = blockIdx.x, h = blockIdx.y;
  const int TP = (T + 15) & ~15;
  const int KP = (T + 31) & ~31;
  const int qoff = h * DHh, koff = Hh + h * DHh, voff = 2 * Hh + h * DHh;

  // stage Q,K (zero-padded to TP rows); 16B chunks
  for (int idx = threadIdx.x; idx < TP * (DHh / 8); idx += 256) {
    int t = idx >> 3;
    int cg = (idx & 7) * 8;
    if (t < T) {
      const bf16* p = QKV + (size_t)(span * rps + t) * QKVW;
      *(v8bf*)&Qs[t * 64 + cg] = *(const v8bf*)(p + qoff + cg);
      *(v8bf*)&Ks[t * 64 + cg] = *(const v8bf*)(p + koff + cg);
    } else {
      v8bf z;
#pragma unroll
      for (int i = 0; i < 8; ++i) z[i] = (bf16)0.f;
      *(v8bf*)&Qs[t * 64 + cg] = z;
      *(v8bf*)&Ks[t * 64 + cg] = z;
    }
  }
  // stage V transposed: Vt[n][k] = V[k][n], zero-padded to KP
  for (int idx = threadIdx.x; idx < 64 * KP; idx += 256) {
    int n = idx / KP, k = idx % KP;
    bf16 v = (bf16)0.f;
    if (k < T) v = QKV[(size_t)(span * rps + k) * QKVW + voff + n];
    Vt[n * KM + k] = v;
  }
  __syncthreads();

  const int w = threadIdx.x >> 5, lane = threadIdx.x & 31;
  const int lr = lane & 15, half = lane >> 4;
  const int nmt = TP >> 4;

  // scores = (Q @ K.T) / 8, column-masked
  for (int tile = w; tile < nmt * nmt; tile += 8) {
    int tm = tile / nmt, tn = tile % nmt;
    v8f c = zero8();
#pragma unroll
    for (int ks = 0; ks < DHh; ks += 32) {
      v16bf a = frag_a_ld(&Qs[(tm * 16 + lr) * 64 + ks]);
      v16bf b = frag_b_ld(&Ks[(tn * 16 + lr) * 64 + ks]);
      c = wmma_bf16(a, b, c);
    }
    int n = tn * 16 + lr;
#pragma unroll
    for (int i = 0; i < 8; ++i) {
      int m = tm * 16 + i + half * 8;
      Ssc[m * SP + n] = (n < T) ? c[i] * 0.125f : -1e30f;
    }
  }
  __syncthreads();

  // row softmax -> Ps (bf16), zero pad rows >= T and cols >= TP
  for (int row = w; row < TP; row += 8) {
    if (row >= T) {
      for (int n = lane; n < KP; n += 32) Ps[row * KM + n] = (bf16)0.f;
      continue;
    }
    float mx = -1e30f;
    for (int n = lane; n < TP; n += 32) mx = fmaxf(mx, Ssc[row * SP + n]);
    mx = wave_max(mx);
    float sum = 0.f;
    for (int n = lane; n < TP; n += 32) sum += __expf(Ssc[row * SP + n] - mx);
    sum = wave_sum(sum);
    float invs = 1.f / sum;
    for (int n = lane; n < KP; n += 32) {
      float p = (n < TP) ? __expf(Ssc[row * SP + n] - mx) * invs : 0.f;
      Ps[row * KM + n] = (bf16)p;
    }
  }
  __syncthreads();

  // O = P @ V   (B fragment reads Vt rows: column n of V, K-contiguous)
  for (int tile = w; tile < nmt * 4; tile += 8) {
    int tm = tile >> 2, tn = tile & 3;
    v8f c = zero8();
    for (int ks = 0; ks < KP; ks += 32) {
      v16bf a = frag_a_ld(&Ps[(tm * 16 + lr) * KM + ks]);
      v16bf b = frag_b_ld(&Vt[(tn * 16 + lr) * KM + ks]);
      c = wmma_bf16(a, b, c);
    }
    int n = tn * 16 + lr;
#pragma unroll
    for (int i = 0; i < 8; ++i) {
      int t = tm * 16 + i + half * 8;
      if (t < T)
        Oout[(size_t)(span * rps + t) * Hh + h * DHh + n] = (bf16)c[i];
    }
  }
}

// ---------------- LayerNorm kernels ----------------------------------------

// out_emb = LN(raw + 2*res) -> f32 + bf16 copies. One block per token.
__global__ void ln_merge_kernel(const float* __restrict__ raw,
                                const bf16* __restrict__ res,
                                const float* __restrict__ g,
                                const float* __restrict__ bb,
                                float* __restrict__ outf,
                                bf16* __restrict__ outh) {
  int m = blockIdx.x;
  __shared__ float xs[Hh];
  __shared__ float rs[16];
  __shared__ float stats[2];
  int t = threadIdx.x, w = t >> 5, lane = t & 31;
  float s = 0.f, q = 0.f;
  for (int k = t; k < Hh; k += 256) {
    float v = raw[(size_t)m * Hh + k] + 2.f * (float)res[(size_t)m * Hh + k];
    xs[k] = v;
    s += v;
    q += v * v;
  }
  s = wave_sum(s);
  q = wave_sum(q);
  if (lane == 0) { rs[w] = s; rs[8 + w] = q; }
  __syncthreads();
  if (t == 0) {
    float a = 0.f, c = 0.f;
    for (int i = 0; i < 8; ++i) { a += rs[i]; c += rs[8 + i]; }
    float mean = a / Hh;
    float var = c / Hh - mean * mean;
    stats[0] = mean;
    stats[1] = rsqrtf(var + 1e-12f);
  }
  __syncthreads();
  float mean = stats[0], rstd = stats[1];
  for (int k = t; k < Hh; k += 256) {
    float y = (xs[k] - mean) * rstd * g[k] + bb[k];
    outf[(size_t)m * Hh + k] = y;
    outh[(size_t)m * Hh + k] = (bf16)y;
  }
}

// out = LN(emb + ff) @ Wout.T + bout. One block per token; 9-way head done
// with LDS float atomics (ds_add_f32).
__global__ void ln_head_kernel(const float* __restrict__ emb,
                               const bf16* __restrict__ ff,
                               const float* __restrict__ g,
                               const float* __restrict__ bb,
                               const float* __restrict__ Wo,
                               const float* __restrict__ bo,
                               float* __restrict__ out) {
  int m = blockIdx.x;
  __shared__ float xs[Hh];
  __shared__ float rs[16];
  __shared__ float stats[2];
  __shared__ float acc[NLl];
  int t = threadIdx.x, w = t >> 5, lane = t & 31;
  if (t < NLl) acc[t] = 0.f;
  float s = 0.f, q = 0.f;
  for (int k = t; k < Hh; k += 256) {
    float v = emb[(size_t)m * Hh + k] + (float)ff[(size_t)m * Hh + k];
    xs[k] = v;
    s += v;
    q += v * v;
  }
  s = wave_sum(s);
  q = wave_sum(q);
  if (lane == 0) { rs[w] = s; rs[8 + w] = q; }
  __syncthreads();
  if (t == 0) {
    float a = 0.f, c = 0.f;
    for (int i = 0; i < 8; ++i) { a += rs[i]; c += rs[8 + i]; }
    float mean = a / Hh;
    float var = c / Hh - mean * mean;
    stats[0] = mean;
    stats[1] = rsqrtf(var + 1e-12f);
  }
  __syncthreads();
  float mean = stats[0], rstd = stats[1];
  float part[NLl];
#pragma unroll
  for (int o = 0; o < NLl; ++o) part[o] = 0.f;
  for (int k = t; k < Hh; k += 256) {
    float y = (xs[k] - mean) * rstd * g[k] + bb[k];
#pragma unroll
    for (int o = 0; o < NLl; ++o) part[o] += y * Wo[(size_t)o * Hh + k];
  }
#pragma unroll
  for (int o = 0; o < NLl; ++o) atomicAdd(&acc[o], part[o]);
  __syncthreads();
  if (t < NLl) out[(size_t)m * NLl + t] = acc[t] + bo[t];
}

// ---------------- host-side orchestration ----------------------------------

extern "C" void kernel_launch(void* const* d_in, const int* in_sizes, int n_in,
                              void* d_out, int out_size, void* d_ws, size_t ws_size,
                              hipStream_t stream) {
  (void)in_sizes; (void)n_in; (void)out_size; (void)ws_size;

  const float* we     = (const float*)d_in[0];
  // d_in[1] = span_starts: fixed contiguous arange(NSP)*SPAN -> implicit here.
  const int*   tags   = (const int*)d_in[2];
  const float* temb   = (const float*)d_in[3];
  const float* saWqkv = (const float*)d_in[4];
  const float* sabqkv = (const float*)d_in[5];
  const float* saWo   = (const float*)d_in[6];
  const float* sabo   = (const float*)d_in[7];
  const float* atWqkv = (const float*)d_in[8];
  const float* atbqkv = (const float*)d_in[9];
  const float* atWo   = (const float*)d_in[10];
  const float* atbo   = (const float*)d_in[11];
  const float* ang    = (const float*)d_in[12];
  const float* anb    = (const float*)d_in[13];
  const float* W1     = (const float*)d_in[14];
  const float* b1     = (const float*)d_in[15];
  const float* W2     = (const float*)d_in[16];
  const float* b2     = (const float*)d_in[17];
  const float* fng    = (const float*)d_in[18];
  const float* fnb    = (const float*)d_in[19];
  const float* Wout   = (const float*)d_in[20];
  const float* bout   = (const float*)d_in[21];
  float* out = (float*)d_out;

  char* p = (char*)d_ws;
  auto alloc = [&](size_t bytes) -> char* {
    char* r = p;
    p += (bytes + 255) & ~(size_t)255;
    return r;
  };

  bf16* xe      = (bf16*)alloc((size_t)M_TOK * Hh * 2);
  bf16* saWqkvH = (bf16*)alloc((size_t)QKVW * Hh * 2);
  bf16* saWoH   = (bf16*)alloc((size_t)Hh * Hh * 2);
  bf16* atWqkvH = (bf16*)alloc((size_t)QKVW * Hh * 2);
  bf16* atWoH   = (bf16*)alloc((size_t)Hh * Hh * 2);
  bf16* W1H     = (bf16*)alloc((size_t)I_FF * Hh * 2);
  bf16* W2H     = (bf16*)alloc((size_t)Hh * I_FF * 2);
  bf16* qkvbuf  = (bf16*)alloc((size_t)M_AT * QKVW * 2);   // reused by ffh
  bf16* attout  = (bf16*)alloc((size_t)M_AT * Hh * 2);
  bf16* Xat     = (bf16*)alloc((size_t)M_AT * Hh * 2);
  bf16* resb    = (bf16*)alloc((size_t)M_TOK * Hh * 2);
  float* embf   = (float*)alloc((size_t)M_TOK * Hh * 4);
  bf16* embh    = (bf16*)alloc((size_t)M_TOK * Hh * 2);
  bf16* ffh     = qkvbuf;  // 100.7 MB overlays qkv+attout+Xat (127.8 MB, dead)

  auto cv = [&](const float* src, bf16* dst, int n) {
    f2bf_kernel<<<(n + 255) / 256, 256, 0, stream>>>(src, dst, n);
  };
  cv(saWqkv, saWqkvH, QKVW * Hh);
  cv(saWo, saWoH, Hh * Hh);
  cv(atWqkv, atWqkvH, QKVW * Hh);
  cv(atWo, atWoH, Hh * Hh);
  cv(W1, W1H, I_FF * Hh);
  cv(W2, W2H, Hh * I_FF);

  {
    size_t pairs = (size_t)M_TOK * (Hh / 2);
    rope_kernel<<<(unsigned)((pairs + 255) / 256), 256, 0, stream>>>(we, xe);
  }
  tag_fill_kernel<<<NSPANS, 256, 0, stream>>>(tags, temb, Xat);

  auto gemm = [&](const bf16* A, const bf16* W, const float* bias,
                  const bf16* resid, bf16* O, int M, int N, int K,
                  int relu, int remap) {
    dim3 grid(N / 128, M / 128);
    gemm_bf16_kernel<<<grid, 256, 0, stream>>>(A, W, bias, resid, O, M, N, K,
                                               relu, remap);
  };

  // ---- sa phase: X = spans + MHA(spans) -> Xat slots 1..64 ----
  gemm(xe, saWqkvH, sabqkv, nullptr, qkvbuf, M_TOK, QKVW, Hh, 0, 0);
  span_attn_kernel<<<dim3(NSPANS, NHh), 256, 0, stream>>>(qkvbuf, attout, SPAN, SPAN);
  gemm(attout, saWoH, sabo, xe, Xat, M_TOK, Hh, Hh, 0, 1);

  // ---- at phase: [tag; X] self-attention, drop the tag row ----
  gemm(Xat, atWqkvH, atbqkv, nullptr, qkvbuf, M_AT, QKVW, Hh, 0, 0);
  span_attn_kernel<<<dim3(NSPANS, NHh), 256, 0, stream>>>(qkvbuf, attout, RPS_AT, RPS_AT);
  gemm(attout, atWoH, atbo, nullptr, resb, M_TOK, Hh, Hh, 0, 2);

  // ---- merge + LN ----
  ln_merge_kernel<<<M_TOK, 256, 0, stream>>>(we, resb, ang, anb, embf, embh);

  // ---- FFN ----
  gemm(embh, W1H, b1, nullptr, ffh, M_TOK, I_FF, Hh, 1, 0);
  gemm(ffh, W2H, b2, nullptr, resb, M_TOK, Hh, I_FF, 0, 0);

  // ---- final LN + 9-way head ----
  ln_head_kernel<<<M_TOK, 256, 0, stream>>>(embf, resb, fng, fnb, Wout, bout, out);
}
